// SubGraphLayer_26989574488250
// MI455X (gfx1250) — matrix operand order repaired and checked by reading, compiled, not run on hardware
//
#include <hip/hip_runtime.h>

typedef _Float16 v2h  __attribute__((ext_vector_type(2)));
typedef __fp16   v2fp __attribute__((ext_vector_type(2)));
typedef _Float16 v16h __attribute__((ext_vector_type(16)));
typedef float    v8f  __attribute__((ext_vector_type(8)));

#define LEN_     128
#define HID_     64
#define NB_      256
#define NN_      2048
#define LN_EPS_  1e-3f
#define NEG_FILL_ (-1.0e9f)

// Order-monotone mapping f32 -> u32 so integer max == float max.
__device__ __forceinline__ unsigned encf(float f) {
  unsigned u = __float_as_uint(f);
  return (u & 0x80000000u) ? ~u : (u | 0x80000000u);
}
__device__ __forceinline__ float decf(unsigned u) {
  unsigned b = (u & 0x80000000u) ? (u & 0x7FFFFFFFu) : ~u;
  return __uint_as_float(b);
}

__device__ __forceinline__ v2h pk_f16(float lo, float hi) {
  v2fp t = __builtin_amdgcn_cvt_pkrtz(lo, hi);
  return __builtin_bit_cast(v2h, t);
}

__global__ __launch_bounds__(256)
void pool_init_kernel(unsigned* __restrict__ pool) {
  int i = blockIdx.x * 256 + threadIdx.x;
  pool[i] = encf(NEG_FILL_);
}

__global__ __launch_bounds__(256)
void mlp_kernel(const float* __restrict__ in, const unsigned char* __restrict__ mask,
                const float* __restrict__ W1, const float* __restrict__ b1,
                const float* __restrict__ W2, const float* __restrict__ b2,
                const float* __restrict__ gamma, const float* __restrict__ beta,
                float* __restrict__ out, unsigned* __restrict__ pool) {
  // Transposed, bank-padded f16 weights + per-wave h1 staging + WG pool max
  __shared__ _Float16 w1t[64 * 132];    // w1t[n][k]  (n=0..63,  k=0..127), stride 132
  __shared__ _Float16 w2t[128 * 68];    // w2t[n][k]  (n=0..127, k=0..63),  stride 68
  __shared__ _Float16 h1s[8][16 * 68];  // per-wave [M=16][K=64], stride 68
  __shared__ unsigned wgmax[128];

  const int tid   = threadIdx.x;
  const int lane  = tid & 31;
  const int wv    = tid >> 5;
  const int laneM = lane & 15;
  const int half  = lane >> 4;

  if (tid < 128) wgmax[tid] = encf(NEG_FILL_);

  // Stage weights transposed (coalesced global reads, padded LDS writes)
  for (int i = tid; i < 128 * 64; i += 256) {
    int k = i >> 6, n = i & 63;                 // W1[k][n]
    w1t[n * 132 + k] = (_Float16)W1[i];
  }
  for (int i = tid; i < 64 * 128; i += 256) {
    int k = i >> 7, n = i & 127;                // W2[k][n]
    w2t[n * 68 + k] = (_Float16)W2[i];
  }
  __syncthreads();

  const size_t tok0 = (size_t)blockIdx.x * 128 + (size_t)wv * 16;
  const float* xrow = in + (tok0 + (size_t)laneM) * LEN_;

  union Frag { v16h v; v2h p[8]; };

  // ---------------- GEMM1: h1[16x64] = x[16x128] @ W1[128x64] ----------------
  v8f acc1[4];
#pragma unroll
  for (int t = 0; t < 4; ++t) { v8f z = {}; acc1[t] = z; }

#pragma unroll
  for (int kb = 0; kb < 4; ++kb) {
    Frag a;
#pragma unroll
    for (int v = 0; v < 8; ++v) {
      int K = kb * 32 + ((v >> 2) << 4) + (half << 3) + ((v & 3) << 1);
      float2 f = *(const float2*)(xrow + K);
      a.p[v] = pk_f16(f.x, f.y);
    }
#pragma unroll
    for (int t = 0; t < 4; ++t) {
      Frag bfr;
      int ncol = t * 16 + laneM;
#pragma unroll
      for (int v = 0; v < 8; ++v) {
        int K = kb * 32 + ((v >> 2) << 4) + (half << 3) + ((v & 3) << 1);
        bfr.p[v] = *(const v2h*)&w1t[ncol * 132 + K];
      }
      acc1[t] = __builtin_amdgcn_wmma_f32_16x16x32_f16(
          false, a.v, false, bfr.v, (short)0, acc1[t], false, false);
    }
  }

  // h1 += b1, store f16 to per-wave LDS tile
#pragma unroll
  for (int t = 0; t < 4; ++t) {
    float bias = b1[t * 16 + laneM];
#pragma unroll
    for (int r = 0; r < 8; ++r) {
      float h = acc1[t][r] + bias;
      h1s[wv][(r + half * 8) * 68 + t * 16 + laneM] = (_Float16)h;
    }
  }
  __syncthreads();

  // ---------------- GEMM2: h2[16x128] = h1[16x64] @ W2[64x128] ----------------
  v8f acc2[8];
#pragma unroll
  for (int t = 0; t < 8; ++t) { v8f z = {}; acc2[t] = z; }

#pragma unroll
  for (int kb = 0; kb < 2; ++kb) {
    Frag a;
#pragma unroll
    for (int v = 0; v < 8; ++v) {
      int K = kb * 32 + ((v >> 2) << 4) + (half << 3) + ((v & 3) << 1);
      a.p[v] = *(const v2h*)&h1s[wv][laneM * 68 + K];
    }
#pragma unroll
    for (int t = 0; t < 8; ++t) {
      Frag bfr;
      int ncol = t * 16 + laneM;
#pragma unroll
      for (int v = 0; v < 8; ++v) {
        int K = kb * 32 + ((v >> 2) << 4) + (half << 3) + ((v & 3) << 1);
        bfr.p[v] = *(const v2h*)&w2t[ncol * 68 + K];
      }
      acc2[t] = __builtin_amdgcn_wmma_f32_16x16x32_f16(
          false, a.v, false, bfr.v, (short)0, acc2[t], false, false);
    }
  }

  // + b2
#pragma unroll
  for (int t = 0; t < 8; ++t) {
    float bias = b2[t * 16 + laneM];
#pragma unroll
    for (int r = 0; r < 8; ++r) acc2[t][r] += bias;
  }

  // LayerNorm stats per row (row M = r + 8*half lives in one 16-lane half)
  float mu[8], inv[8];
#pragma unroll
  for (int r = 0; r < 8; ++r) {
    float s = 0.f, q = 0.f;
#pragma unroll
    for (int t = 0; t < 8; ++t) { float x = acc2[t][r]; s += x; q += x * x; }
#pragma unroll
    for (int m = 1; m <= 8; m <<= 1) {
      s += __shfl_xor(s, m, 32);
      q += __shfl_xor(q, m, 32);
    }
    float m1 = s * (1.0f / 128.0f);
    float m2 = q * (1.0f / 128.0f);
    mu[r]  = m1;
    inv[r] = rsqrtf(m2 - m1 * m1 + LN_EPS_);
  }

  float g[8], be[8];
#pragma unroll
  for (int t = 0; t < 8; ++t) {
    int col = t * 16 + laneM;
    g[t] = gamma[col]; be[t] = beta[col];
  }

  float cmax[8];
#pragma unroll
  for (int t = 0; t < 8; ++t) cmax[t] = NEG_FILL_;

#pragma unroll
  for (int r = 0; r < 8; ++r) {
    size_t tk = tok0 + (size_t)(half * 8 + r);
    bool mk = mask[tk] != 0;
    float* orow = out + tk * 256;
#pragma unroll
    for (int t = 0; t < 8; ++t) {
      float x = (acc2[t][r] - mu[r]) * inv[r] * g[t] + be[t];
      x = fmaxf(x, 0.0f);
      x = mk ? x : NEG_FILL_;
      orow[t * 16 + laneM] = x;
      cmax[t] = fmaxf(cmax[t], x);
    }
  }

  // wave column max -> WG LDS max -> per-batch global max (encoded u32)
#pragma unroll
  for (int t = 0; t < 8; ++t) {
    float m = fmaxf(cmax[t], __shfl_xor(cmax[t], 16, 32));
    if (half == 0) atomicMax(&wgmax[t * 16 + laneM], encf(m));
  }
  __syncthreads();
  if (tid < 128) {
    int b = blockIdx.x >> 4;  // 2048/128 = 16 workgroups per batch
    atomicMax(&pool[b * 128 + tid], wgmax[tid]);
  }
}

__global__ __launch_bounds__(256)
void pool_bcast_kernel(const unsigned* __restrict__ pool, float* __restrict__ out) {
  // block = 8 rows, thread = 4 columns
  size_t row = (size_t)blockIdx.x * 8 + (threadIdx.x >> 5);
  int c = (threadIdx.x & 31) * 4;
  int b = (int)(row >> 11);  // N = 2048
  const unsigned* p = pool + b * 128 + c;
  float4 v = make_float4(decf(p[0]), decf(p[1]), decf(p[2]), decf(p[3]));
  *(float4*)(out + row * 256 + 128 + c) = v;
}

extern "C" void kernel_launch(void* const* d_in, const int* in_sizes, int n_in,
                              void* d_out, int out_size, void* d_ws, size_t ws_size,
                              hipStream_t stream) {
  const float*         in    = (const float*)d_in[0];
  const unsigned char* mask  = (const unsigned char*)d_in[1];  // jnp bool -> 1 byte
  const float*         W1    = (const float*)d_in[2];
  const float*         b1    = (const float*)d_in[3];
  const float*         W2    = (const float*)d_in[4];
  const float*         b2    = (const float*)d_in[5];
  const float*         gamma = (const float*)d_in[6];
  const float*         beta  = (const float*)d_in[7];
  float*    out  = (float*)d_out;
  unsigned* pool = (unsigned*)d_ws;  // NB_*LEN_ u32 = 128 KB

  hipLaunchKernelGGL(pool_init_kernel, dim3((NB_ * LEN_) / 256), dim3(256), 0, stream, pool);
  hipLaunchKernelGGL(mlp_kernel, dim3((NB_ * NN_) / 128), dim3(256), 0, stream,
                     in, mask, W1, b1, W2, b2, gamma, beta, out, pool);
  hipLaunchKernelGGL(pool_bcast_kernel, dim3((NB_ * NN_) / 8), dim3(256), 0, stream, pool, out);
}
